// CostVolume2DAggregation_4063039062166
// MI455X (gfx1250) — matrix operand order repaired and checked by reading, not compiled, not run on hardware
//
#include <hip/hip_runtime.h>

// ---------------------------------------------------------------------------
// CostVolume2DAggregation for MI455X (gfx1250, wave32, WMMA)
//   cost volume (f16) -> conv3x3 (192->64) WMMA implicit GEMM -> BN+ReLU
//   -> conv3x3 (64->32) WMMA -> BN+ReLU -> ConvTranspose2d(32->1,4,2,1)
// Each wave: 2 M-tiles (32-pixel strip) x 16 output channels, so every
// B fragment (2x global_load_b128, L2-resident) is reused twice from regs;
// A fragments are 2x ds_load_b128 from a channel-major LDS halo row.
// All three fragment loads are issued before either WMMA to keep both
// matrix ops back-to-back in the XDL pipe.
// ---------------------------------------------------------------------------

typedef __attribute__((ext_vector_type(16))) _Float16 v16h;
typedef __attribute__((ext_vector_type(8)))  _Float16 v8h;
typedef __attribute__((ext_vector_type(4)))  _Float16 v4h;
typedef __attribute__((ext_vector_type(8)))  float    v8f;

#define BB 8
#define CF 32        // input feature channels
#define HH 128
#define WW 256
#define DD 192       // MAX_DISP
#define C1 64
#define C2 32
#define HW (HH * WW)
#define BN_EPS 1e-5f
#define BN_PARTS 32  // partial-reduction blocks per channel

// ---------------------------------------------------------------------------
// 1) Cost volume: cost[b,d,h,w] = mean_c |lf[b,c,h,w]-rf[b,c,h,w-d]| (w>=d)
//    4 outputs per thread; lf read as float4, rf clamped scalar (no OOB).
// ---------------------------------------------------------------------------
__global__ __launch_bounds__(256)
void cost_kernel(const float* __restrict__ lf, const float* __restrict__ rf,
                 _Float16* __restrict__ cost) {
  long idx = (long)blockIdx.x * blockDim.x + threadIdx.x;
  const long total4 = (long)BB * DD * HH * (WW / 4);
  if (idx >= total4) return;
  int w4 = (int)(idx % (WW / 4)); long t = idx / (WW / 4);
  int h  = (int)(t % HH);         t /= HH;
  int d  = (int)(t % DD);         int b = (int)(t / DD);
  const int w0 = w4 * 4;

  const float* lp = lf + (long)b * CF * HW + h * WW + w0;
  const float* rp = rf + (long)b * CF * HW + h * WW;   // row base

  float s0 = 0.f, s1 = 0.f, s2 = 0.f, s3 = 0.f;
  const int c0 = max(w0 + 0 - d, 0), c1s = max(w0 + 1 - d, 0);
  const int c2s = max(w0 + 2 - d, 0), c3s = max(w0 + 3 - d, 0);
  const float m0 = (w0 + 0 >= d) ? 1.f : 0.f, m1 = (w0 + 1 >= d) ? 1.f : 0.f;
  const float m2 = (w0 + 2 >= d) ? 1.f : 0.f, m3 = (w0 + 3 >= d) ? 1.f : 0.f;
#pragma unroll 4
  for (int c = 0; c < CF; ++c) {
    float4 lv = *(const float4*)(lp + (long)c * HW);
    const float* rpc = rp + (long)c * HW;
    s0 += fabsf(lv.x - rpc[c0]);
    s1 += fabsf(lv.y - rpc[c1s]);
    s2 += fabsf(lv.z - rpc[c2s]);
    s3 += fabsf(lv.w - rpc[c3s]);
  }
  const float k = 1.0f / CF;
  v4h o;
  o[0] = (_Float16)(s0 * m0 * k); o[1] = (_Float16)(s1 * m1 * k);
  o[2] = (_Float16)(s2 * m2 * k); o[3] = (_Float16)(s3 * m3 * k);
  *(v4h*)(cost + idx * 4) = o;
}

// ---------------------------------------------------------------------------
// 2) Weight conversion: OIHW f32 -> WMMA B-fragment layout f16.
//    idx = (((kh*3+kw)*CB + cb)*NT + nt)*512 + lane*16 + i
//    lane: n = lane&15, khalf = lane>>4;  kk = i<8 ? khalf*8 + i : 16 + khalf*8 + (i-8)
//    element = w[co = nt*16+n][ci = cb*32+kk][kh][kw]
// ---------------------------------------------------------------------------
__global__ __launch_bounds__(256)
void wcvt_frag_kernel(const float* __restrict__ w, _Float16* __restrict__ out,
                      int O, int I) {
  const int NT = O / 16, CB = I / 32;
  int idx = blockIdx.x * blockDim.x + threadIdx.x;
  if (idx >= 9 * CB * NT * 512) return;
  int i    = idx & 15;
  int lane = (idx >> 4) & 31;
  int nt   = (idx >> 9) % NT;
  int cb   = (idx / (512 * NT)) % CB;
  int kk9  = idx / (512 * NT * CB);          // kh*3+kw
  int n = lane & 15, khalf = lane >> 4;
  int kk = (i < 8) ? (khalf * 8 + i) : (16 + khalf * 8 + (i - 8));
  int co = nt * 16 + n;
  int ci = cb * 32 + kk;
  int kh = kk9 / 3, kw = kk9 % 3;
  out[idx] = (_Float16)w[((co * I + ci) * 3 + kh) * 3 + kw];
}

// ---------------------------------------------------------------------------
// 3) 3x3 conv as 9 shifted 1x1 GEMMs, K-chunked by 32, WMMA f32<-f16.
//    Block = (COUT/16) waves over a 32-pixel M strip (2 accumulators/wave).
// ---------------------------------------------------------------------------
template <int CIN, int COUT>
__global__ __launch_bounds__(32 * (COUT / 16))
void conv3x3_wmma_kernel(const _Float16* __restrict__ in,    // [B][CIN][H][W]
                         const _Float16* __restrict__ wfrag, // fragment layout
                         const float* __restrict__ bias,
                         float* __restrict__ out) {          // [B][COUT][H][W]
  constexpr int NT = COUT / 16;
  constexpr int CB = CIN / 32;
  __shared__ alignas(16) _Float16 a_lds[34 * CIN];   // [dw][c], channel-major

  const int tid      = threadIdx.x;
  const int lane     = tid & 31;
  const int wave     = tid >> 5;            // N-tile index
  const int nthreads = 32 * NT;

  int bid = blockIdx.x;
  const int wt = bid % (WW / 32); bid /= (WW / 32);
  const int h  = bid % HH;        bid /= HH;
  const int b  = bid;
  const int w0 = wt * 32;

  const int m     = lane & 15;              // M row (A) / N col (B,C)
  const int khalf = lane >> 4;              // K-half selector per ISA layout

  v8f acc0 = {};                            // pixels w0 + m
  v8f acc1 = {};                            // pixels w0 + 16 + m

  for (int kh = 0; kh < 3; ++kh) {
    const int  hin = h + kh - 1;
    const bool hok = (hin >= 0) && (hin < HH);
    __syncthreads();                        // prior a_lds reads done
    // Stage halo strip transposed: a_lds[dw*CIN + c] = in[b][c][hin][w0-1+dw]
    for (int i = tid; i < 34 * CIN; i += nthreads) {
      int c = i / 34, dw = i % 34;
      int win = w0 - 1 + dw;
      _Float16 v = (_Float16)0.f;
      if (hok && (unsigned)win < (unsigned)WW)
        v = in[((long)b * CIN + c) * HW + hin * WW + win];
      a_lds[dw * CIN + c] = v;
    }
    if (hok && hin + 1 < HH && tid == 0)    // global_prefetch_b8 next row
      __builtin_prefetch((const void*)(in + (long)b * CIN * HW + (hin + 1) * WW + w0), 0, 1);
    __syncthreads();                        // staging visible

#pragma unroll
    for (int kw = 0; kw < 3; ++kw) {
#pragma unroll
      for (int cb = 0; cb < CB; ++cb) {
        // Issue all three fragment loads first, then both WMMAs back-to-back.
        // B fragment: 32 contiguous bytes per lane, coalesced; reused 2x.
        const _Float16* wp = wfrag +
            ((long)(((kh * 3 + kw) * CB + cb) * NT + wave)) * 512 + lane * 16;
        v16h bfrag = *(const v16h*)wp;
        // A fragments: lane (m,khalf) reads K runs khalf*8.. and 16+khalf*8..
        // at columns dw = m+kw and dw = m+16+kw -> 2x ds_load_b128 each.
        const int ab0 = (m + kw) * CIN + cb * 32 + khalf * 8;
        v8h a0lo = *(const v8h*)(a_lds + ab0);
        v8h a0hi = *(const v8h*)(a_lds + ab0 + 16);
        const int ab1 = (m + 16 + kw) * CIN + cb * 32 + khalf * 8;
        v8h a1lo = *(const v8h*)(a_lds + ab1);
        v8h a1hi = *(const v8h*)(a_lds + ab1 + 16);
        v16h af0 = __builtin_shufflevector(a0lo, a0hi,
            0, 1, 2, 3, 4, 5, 6, 7, 8, 9, 10, 11, 12, 13, 14, 15);
        v16h af1 = __builtin_shufflevector(a1lo, a1hi,
            0, 1, 2, 3, 4, 5, 6, 7, 8, 9, 10, 11, 12, 13, 14, 15);
        acc0 = __builtin_amdgcn_wmma_f32_16x16x32_f16(
            false, af0, false, bfrag, (short)0, acc0, false, false);
        acc1 = __builtin_amdgcn_wmma_f32_16x16x32_f16(
            false, af1, false, bfrag, (short)0, acc1, false, false);
      }
    }
  }

  // C/D layout: VGPR j -> M = j + 8*khalf, N = lane&15; 8 contiguous floats.
  const int   co = wave * 16 + m;
  const float bv = bias[co];
  float* op = out + ((long)b * COUT + co) * HW + h * WW + w0 + 8 * khalf;
  *(float4*)(op)          = make_float4(acc0[0] + bv, acc0[1] + bv, acc0[2] + bv, acc0[3] + bv);
  *(float4*)(op + 4)      = make_float4(acc0[4] + bv, acc0[5] + bv, acc0[6] + bv, acc0[7] + bv);
  *(float4*)(op + 16)     = make_float4(acc1[0] + bv, acc1[1] + bv, acc1[2] + bv, acc1[3] + bv);
  *(float4*)(op + 16 + 4) = make_float4(acc1[4] + bv, acc1[5] + bv, acc1[6] + bv, acc1[7] + bv);
}

// ---------------------------------------------------------------------------
// 4) Training-mode BN statistics, three stages for full-chip parallelism:
//    zero -> partial block reductions + global_atomic_add_f32 -> finalize.
//    stats layout: [0,C) sum  [C,2C) sumsq  [2C,3C) mean  [3C,4C) rstd
// ---------------------------------------------------------------------------
__global__ __launch_bounds__(256)
void zero_kernel(float* __restrict__ p, int n) {
  int i = blockIdx.x * blockDim.x + threadIdx.x;
  if (i < n) p[i] = 0.f;
}

__global__ __launch_bounds__(256)
void bnpartial_kernel(const float* __restrict__ y, float* __restrict__ stats, int C) {
  __shared__ float s_sum[256], s_sq[256];
  const int c = blockIdx.x / BN_PARTS, part = blockIdx.x % BN_PARTS;
  const int tid = threadIdx.x;
  const long n = (long)BB * HW, chunk = n / BN_PARTS;
  float s = 0.f, sq = 0.f;
  for (long i = part * chunk + tid; i < (part + 1) * chunk; i += 256) {
    long bidx = i / HW, hw = i % HW;
    float v = y[(bidx * C + c) * HW + hw];
    s += v; sq += v * v;
  }
  s_sum[tid] = s; s_sq[tid] = sq;
  __syncthreads();
  for (int off = 128; off > 0; off >>= 1) {
    if (tid < off) { s_sum[tid] += s_sum[tid + off]; s_sq[tid] += s_sq[tid + off]; }
    __syncthreads();
  }
  if (tid == 0) {
    atomicAdd(&stats[c], s_sum[0]);          // global_atomic_add_f32
    atomicAdd(&stats[C + c], s_sq[0]);
  }
}

__global__ __launch_bounds__(64)
void bnfinal_kernel(float* __restrict__ stats, int C) {
  int c = blockIdx.x * blockDim.x + threadIdx.x;
  if (c >= C) return;
  const float n = (float)((long)BB * HW);
  float mm = stats[c] / n;
  float vv = stats[C + c] / n - mm * mm;
  stats[2 * C + c] = mm;
  stats[3 * C + c] = rsqrtf(vv + BN_EPS);
}

// ---------------------------------------------------------------------------
// 5) BN apply + ReLU, 4 elements/thread, optional down-convert to f16.
// ---------------------------------------------------------------------------
__device__ inline void store4(_Float16* p, float a, float b, float c, float d) {
  v4h o; o[0] = (_Float16)a; o[1] = (_Float16)b; o[2] = (_Float16)c; o[3] = (_Float16)d;
  *(v4h*)p = o;
}
__device__ inline void store4(float* p, float a, float b, float c, float d) {
  *(float4*)p = make_float4(a, b, c, d);
}

template <typename T>
__global__ __launch_bounds__(256)
void bnrelu_kernel(const float* __restrict__ y, const float* __restrict__ stats,
                   const float* __restrict__ g, const float* __restrict__ bta,
                   T* __restrict__ x, int C) {
  long idx4 = (long)blockIdx.x * blockDim.x + threadIdx.x;
  if (idx4 >= (long)BB * C * HW / 4) return;
  int c = (int)((idx4 * 4 / HW) % C);        // HW % 4 == 0: one channel per quad
  const float mm = stats[2 * C + c], rs = stats[3 * C + c];
  const float sc = rs * g[c], sb = bta[c] - mm * sc;
  float4 v = ((const float4*)y)[idx4];
  float a0 = fmaxf(v.x * sc + sb, 0.f), a1 = fmaxf(v.y * sc + sb, 0.f);
  float a2 = fmaxf(v.z * sc + sb, 0.f), a3 = fmaxf(v.w * sc + sb, 0.f);
  store4(x + idx4 * 4, a0, a1, a2, a3);
}

// ---------------------------------------------------------------------------
// 6) ConvTranspose2d(32,1,4,2,1): each output sees 2x2 valid taps by parity.
// ---------------------------------------------------------------------------
__global__ __launch_bounds__(256)
void deconv_kernel(const float* __restrict__ x,    // [B][32][H][W]
                   const float* __restrict__ wts,  // [32][1][4][4]
                   const float* __restrict__ bias,
                   float* __restrict__ out) {      // [B][1][2H][2W]
  const int OH = 2 * HH, OW = 2 * WW;
  long idx = (long)blockIdx.x * blockDim.x + threadIdx.x;
  if (idx >= (long)BB * OH * OW) return;
  int ow = (int)(idx % OW); long t = idx / OW;
  int oh = (int)(t % OH);   int b = (int)(t / OH);
  float s = bias[0];
  const int kh0 = (oh + 1) & 1, kw0 = (ow + 1) & 1;
  for (int dk = 0; dk < 2; ++dk) {
    int kh = kh0 + 2 * dk;
    int ih = (oh + 1 - kh) >> 1;
    if (ih < 0 || ih >= HH) continue;
    for (int dj = 0; dj < 2; ++dj) {
      int kw = kw0 + 2 * dj;
      int iw = (ow + 1 - kw) >> 1;
      if (iw < 0 || iw >= WW) continue;
      const float* xp = x + (long)b * C2 * HW + ih * WW + iw;
#pragma unroll 4
      for (int c = 0; c < C2; ++c) s += xp[c * HW] * wts[(c * 4 + kh) * 4 + kw];
    }
  }
  out[idx] = s;
}

// ---------------------------------------------------------------------------
// Host-side orchestration (all on `stream`, graph-capture safe).
// ---------------------------------------------------------------------------
extern "C" void kernel_launch(void* const* d_in, const int* in_sizes, int n_in,
                              void* d_out, int out_size, void* d_ws, size_t ws_size,
                              hipStream_t stream) {
  (void)in_sizes; (void)n_in; (void)out_size; (void)ws_size;
  const float* lf       = (const float*)d_in[0];
  const float* rf       = (const float*)d_in[1];
  const float* conv1_w  = (const float*)d_in[2];
  const float* conv1_b  = (const float*)d_in[3];
  const float* bn1_g    = (const float*)d_in[4];
  const float* bn1_b    = (const float*)d_in[5];
  const float* conv2_w  = (const float*)d_in[6];
  const float* conv2_b  = (const float*)d_in[7];
  const float* bn2_g    = (const float*)d_in[8];
  const float* bn2_b    = (const float*)d_in[9];
  const float* deconv_w = (const float*)d_in[10];
  const float* deconv_b = (const float*)d_in[11];
  float* out = (float*)d_out;

  char* ws = (char*)d_ws;
  // Workspace layout (bytes); cost region is recycled for y2/x2 after conv1.
  const size_t off_w1   = 0;                                   // 9*6*4*512 f16
  const size_t off_w2   = off_w1 + (size_t)9 * DD * C1 * 2;    // 221184
  const size_t off_st1  = off_w2 + (size_t)9 * C1 * C2 * 2;    // +36864
  const size_t off_st2  = off_st1 + 4 * C1 * 4;                // 4C floats each
  const size_t off_cost = 262144;                              // 256KB aligned
  const size_t off_y1   = off_cost + (size_t)BB * DD * HW * 2; // +96MB
  const size_t off_x1   = off_y1 + (size_t)BB * C1 * HW * 4;   // +64MB
  const size_t off_y2   = off_cost;                            // reuse (cost dead)
  const size_t off_x2   = off_cost + (size_t)BB * C2 * HW * 4;

  _Float16* w1buf = (_Float16*)(ws + off_w1);
  _Float16* w2buf = (_Float16*)(ws + off_w2);
  float*    st1   = (float*)(ws + off_st1);
  float*    st2   = (float*)(ws + off_st2);
  _Float16* cost  = (_Float16*)(ws + off_cost);
  float*    y1    = (float*)(ws + off_y1);
  _Float16* x1    = (_Float16*)(ws + off_x1);
  float*    y2    = (float*)(ws + off_y2);
  float*    x2    = (float*)(ws + off_x2);

  // 1) cost volume (f16), 4 outputs per thread
  {
    long total4 = (long)BB * DD * HH * (WW / 4);
    cost_kernel<<<(int)((total4 + 255) / 256), 256, 0, stream>>>(lf, rf, cost);
  }
  // 2) weight conversion into WMMA B-fragment layout
  wcvt_frag_kernel<<<(9 * DD * C1 + 255) / 256, 256, 0, stream>>>(conv1_w, w1buf, C1, DD);
  wcvt_frag_kernel<<<(9 * C1 * C2 + 255) / 256, 256, 0, stream>>>(conv2_w, w2buf, C2, C1);
  // 3) conv1: 192 -> 64 (WMMA), 4 waves/block, 32-pixel strip
  conv3x3_wmma_kernel<DD, C1><<<BB * HH * (WW / 32), 128, 0, stream>>>(cost, w1buf, conv1_b, y1);
  // 4) BN1 stats (zero -> partial+atomics -> finalize), apply+ReLU -> f16
  zero_kernel<<<1, 256, 0, stream>>>(st1, 2 * C1);
  bnpartial_kernel<<<C1 * BN_PARTS, 256, 0, stream>>>(y1, st1, C1);
  bnfinal_kernel<<<1, 64, 0, stream>>>(st1, C1);
  {
    long total4 = (long)BB * C1 * HW / 4;
    bnrelu_kernel<_Float16><<<(int)((total4 + 255) / 256), 256, 0, stream>>>(y1, st1, bn1_g, bn1_b, x1, C1);
  }
  // 5) conv2: 64 -> 32 (WMMA), 2 waves/block, 32-pixel strip
  conv3x3_wmma_kernel<C1, C2><<<BB * HH * (WW / 32), 64, 0, stream>>>(x1, w2buf, conv2_b, y2);
  // 6) BN2 stats + apply + ReLU -> f32
  zero_kernel<<<1, 256, 0, stream>>>(st2, 2 * C2);
  bnpartial_kernel<<<C2 * BN_PARTS, 256, 0, stream>>>(y2, st2, C2);
  bnfinal_kernel<<<1, 64, 0, stream>>>(st2, C2);
  {
    long total4 = (long)BB * C2 * HW / 4;
    bnrelu_kernel<float><<<(int)((total4 + 255) / 256), 256, 0, stream>>>(y2, st2, bn2_g, bn2_b, x2, C2);
  }
  // 7) transposed conv -> output (B, 1, 256, 512)
  {
    long total = (long)BB * (2 * HH) * (2 * WW);
    deconv_kernel<<<(int)((total + 255) / 256), 256, 0, stream>>>(x2, deconv_w, deconv_b, out);
  }
}
